// Model_51135880626314
// MI455X (gfx1250) — matrix operand order repaired and checked by reading, not compiled, not run on hardware
//
#include <hip/hip_runtime.h>
#include <cmath>
#include <cstddef>
#include <cstdint>

// ---------------------------------------------------------------------------
// Informer forward for MI455X (gfx1250, wave32).
// Dense linears run on v_wmma_f32_16x16x32_f16 with a weight-stationary
// register-resident B matrix (whole weight fits in <=192 VGPRs of fragments),
// activations staged per-wave through LDS. ProbSparse pieces are wave32 VALU
// kernels with LDS + shuffle reductions.
// ---------------------------------------------------------------------------

typedef __attribute__((ext_vector_type(16))) _Float16 v16h;
typedef __attribute__((ext_vector_type(8)))  float    v8f;

#define BATCH   32
#define NH      4
#define DM      64
#define DFF     128
#define SEQ_LEN 720
#define DEC_LEN 672
#define ENC_L2  360
#define PRED    336
#define CIN     7

static inline int ceil_div(int a, int b) { return (a + b - 1) / b; }

// ---------------------------------------------------------------------------
// Weight-stationary WMMA GEMM: Y[M,N] = act( X[M,K] @ W[N,K]^T + bias ).
// Compile-time: KT = K/32 k-tiles, NT = n-tiles (NT*16 >= N).
// Runtime:      M % 16 == 0 (true for all calls), N guarded per column.
// Block = 256 threads = 8 waves; each wave owns independent M-tiles
// (grid-stride), B fragments live in registers, A tile staged in per-wave LDS
// (no block barrier needed: LDS ops are in-order within a wave).
// ---------------------------------------------------------------------------
template <int KT, int NT>
__global__ __launch_bounds__(256)
void gemm_ws_kernel(const float* __restrict__ X, const float* __restrict__ W,
                    const float* __restrict__ bias, float* __restrict__ Y,
                    int M, int N, int act)
{
    constexpr int K = KT * 32;
    __shared__ _Float16 sA[8][16 * 32];   // 8 KB: per-wave A staging

    const int wave = threadIdx.x >> 5;
    const int lane = threadIdx.x & 31;
    const int half = lane >> 4;           // 0: lanes 0..15, 1: lanes 16..31
    const int l16  = lane & 15;

    // ---- B fragments, weight-stationary (identical in every wave) ----
    // B (32x16) layout: lane holds column N=l16;
    //   lanes 0-15: v_r = K{2r,2r+1}; lanes 16-31: K{16+2r,17+2r}
    v16h bf[KT][NT];
    #pragma unroll
    for (int kt = 0; kt < KT; ++kt) {
        #pragma unroll
        for (int nt = 0; nt < NT; ++nt) {
            const int wr = nt * 16 + l16;
            if (wr < N) {
                const float* wsrc = W + (size_t)wr * K + kt * 32 + half * 16;
                #pragma unroll
                for (int r = 0; r < 8; ++r) {
                    bf[kt][nt][2 * r]     = (_Float16)wsrc[2 * r];
                    bf[kt][nt][2 * r + 1] = (_Float16)wsrc[2 * r + 1];
                }
            } else {
                #pragma unroll
                for (int e = 0; e < 16; ++e) bf[kt][nt][e] = (_Float16)0.f;
            }
        }
    }

    const int mtiles = M >> 4;
    for (int mt = blockIdx.x * 8 + wave; mt < mtiles; mt += gridDim.x * 8) {
        const int row0 = mt * 16;
        v8f acc[NT];
        #pragma unroll
        for (int nt = 0; nt < NT; ++nt)
            #pragma unroll
            for (int e = 0; e < 8; ++e) acc[nt][e] = 0.f;

        #pragma unroll
        for (int kt = 0; kt < KT; ++kt) {
            // Stage A tile (16x32) into this wave's LDS slice, coalesced:
            // lane handles row (lane>>1), 16 consecutive K-cols.
            {
                const int r  = lane >> 1;
                const int cb = (lane & 1) * 16;
                const float* xs = X + (size_t)(row0 + r) * K + kt * 32 + cb;
                #pragma unroll
                for (int c = 0; c < 16; ++c)
                    sA[wave][r * 32 + cb + c] = (_Float16)xs[c];
            }
            // A fragment (16x32 f16, contiguous k order, same as B):
            //   lanes 0-15 : M=lane   ; v0..v7 K=0..15
            //   lanes16-31 : M=lane-16; v0..v7 K=16..31
            v16h a;
            #pragma unroll
            for (int r = 0; r < 8; ++r) {
                const int ka = half * 16 + 2 * r;
                a[2 * r]     = sA[wave][l16 * 32 + ka];
                a[2 * r + 1] = sA[wave][l16 * 32 + ka + 1];
            }
            #pragma unroll
            for (int nt = 0; nt < NT; ++nt)
                acc[nt] = __builtin_amdgcn_wmma_f32_16x16x32_f16(
                    false, a, false, bf[kt][nt], (short)0, acc[nt], false, false);
        }

        // C/D layout: VGPR r -> (M = half*8 + r, N = l16)
        #pragma unroll
        for (int nt = 0; nt < NT; ++nt) {
            const int n = nt * 16 + l16;
            if (n < N) {
                const float bv = bias ? bias[n] : 0.f;
                #pragma unroll
                for (int r = 0; r < 8; ++r) {
                    const int m = row0 + half * 8 + r;
                    float v = acc[nt][r] + bv;
                    if (act == 1) // exact GELU
                        v = 0.5f * v * (1.f + erff(v * 0.70710678118f));
                    Y[(size_t)m * N + n] = v;
                }
            }
        }
    }
}

// ---------------------------------------------------------------------------
// Data embedding: circular token conv (Cin->64, k=3) + time linear + pos enc.
// ---------------------------------------------------------------------------
__global__ void embed_kernel(const float* __restrict__ x, const float* __restrict__ xm,
                             const float* __restrict__ tokw, const float* __restrict__ timw,
                             float* __restrict__ out, int L, int Cin)
{
    int idx = blockIdx.x * blockDim.x + threadIdx.x;
    int total = BATCH * L * DM;
    if (idx >= total) return;
    int o = idx & (DM - 1);
    int t = (idx >> 6) % L;
    int b = idx / (DM * L);

    float acc = 0.f;
    #pragma unroll
    for (int kk = 0; kk < 3; ++kk) {
        int ts = t - 1 + kk;
        ts = (ts < 0) ? ts + L : ((ts >= L) ? ts - L : ts);
        const float* xr = x + ((size_t)b * L + ts) * Cin;
        for (int i = 0; i < Cin; ++i)
            acc += xr[i] * tokw[(o * Cin + i) * 3 + kk];
    }
    const float* xmr = xm + ((size_t)b * L + t) * 4;
    #pragma unroll
    for (int j = 0; j < 4; ++j) acc += xmr[j] * timw[o * 4 + j];

    const int   p2  = o >> 1;
    const float dv  = __expf(-(float)(2 * p2) * (9.210340372f / 64.f));
    const float ang = (float)t * dv;
    acc += (o & 1) ? __cosf(ang) : __sinf(ang);
    out[idx] = acc;
}

// ---------------------------------------------------------------------------
// ProbSparse sparsity measure M[l] = max_s(q.k_s) - sum_s(q.k_s)/Lk.
// ---------------------------------------------------------------------------
__global__ void attn_m_kernel(const float* __restrict__ Q, const float* __restrict__ Kb,
                              float* __restrict__ Ms, int Lq, int Lk, int Upart)
{
    int idx = blockIdx.x * blockDim.x + threadIdx.x;
    int total = BATCH * NH * Lq;
    if (idx >= total) return;
    int l = idx % Lq;
    int h = (idx / Lq) & (NH - 1);
    int b = idx / (Lq * NH);

    const float* q = Q + ((size_t)b * Lq + l) * DM + h * 16;
    float qv[16];
    #pragma unroll
    for (int d = 0; d < 16; ++d) qv[d] = q[d];

    float mx = -3.4e38f, sm = 0.f;
    unsigned rng = (unsigned)l * 2654435761u ^ 0x9E3779B9u;
    for (int s = 0; s < Upart; ++s) {
        rng = rng * 1664525u + 1013904223u + (unsigned)s * 0x85EBCA6Bu;
        int ks = (int)(rng % (unsigned)Lk);
        const float* kr = Kb + ((size_t)b * Lk + ks) * DM + h * 16;
        float dot = 0.f;
        #pragma unroll
        for (int d = 0; d < 16; ++d) dot += qv[d] * kr[d];
        mx = fmaxf(mx, dot);
        sm += dot;
    }
    Ms[((size_t)b * NH + h) * Lq + l] = mx - sm / (float)Lk;
}

// ---------------------------------------------------------------------------
// Top-u selection per (b,h): iterative wave32 argmax over LDS copy.
// ---------------------------------------------------------------------------
__global__ __launch_bounds__(32)
void topk_kernel(const float* __restrict__ Ms, int* __restrict__ Mtop, int Lq, int u)
{
    __shared__ float sm[768];
    const int bh = blockIdx.x, lane = threadIdx.x;
    const float* src = Ms + (size_t)bh * Lq;
    for (int i = lane; i < Lq; i += 32) sm[i] = src[i];
    __syncthreads();
    for (int it = 0; it < u; ++it) {
        float best = -3.4e38f; int bi = 0;
        for (int i = lane; i < Lq; i += 32)
            if (sm[i] > best) { best = sm[i]; bi = i; }
        for (int o = 16; o > 0; o >>= 1) {
            float ob  = __shfl_xor(best, o, 32);
            int   obi = __shfl_xor(bi, o, 32);
            if (ob > best || (ob == best && obi < bi)) { best = ob; bi = obi; }
        }
        if (lane == 0) Mtop[(size_t)bh * u + it] = bi;
        sm[bi] = -3.4e38f; // all lanes write identical value
        __syncthreads();
    }
}

// Context init (non-causal): mean of V over keys, then broadcast over queries.
__global__ void mean_v_kernel(const float* __restrict__ V, float* __restrict__ mv, int Lk)
{
    int idx = blockIdx.x * blockDim.x + threadIdx.x;
    if (idx >= BATCH * DM) return;
    int c = idx & (DM - 1), b = idx >> 6;
    float s = 0.f;
    for (int l = 0; l < Lk; ++l) s += V[((size_t)b * Lk + l) * DM + c];
    mv[idx] = s / (float)Lk;
}
__global__ void fill_mean_kernel(const float* __restrict__ mv, float* __restrict__ ctx, int Lq)
{
    int idx = blockIdx.x * blockDim.x + threadIdx.x;
    if (idx >= BATCH * Lq * DM) return;
    int c = idx & (DM - 1);
    int b = idx / (DM * Lq);
    ctx[idx] = mv[b * DM + c];
}
// Context init (causal): running cumsum of V along keys.
__global__ void cumsum_v_kernel(const float* __restrict__ V, float* __restrict__ ctx, int L)
{
    int idx = blockIdx.x * blockDim.x + threadIdx.x;
    if (idx >= BATCH * DM) return;
    int c = idx & (DM - 1), b = idx >> 6;
    float run = 0.f;
    for (int l = 0; l < L; ++l) {
        run += V[((size_t)b * L + l) * DM + c];
        ctx[((size_t)b * L + l) * DM + c] = run;
    }
}

// ---------------------------------------------------------------------------
// Softmax(QK^T/sqrt(d)) @ V for the u selected rows; overwrite ctx rows.
// One wave per (b,h,selected-row); scores staged in LDS.
// ---------------------------------------------------------------------------
__global__ __launch_bounds__(32)
void attn_update_kernel(const float* __restrict__ Q, const float* __restrict__ Kb,
                        const float* __restrict__ V, const int* __restrict__ Mtop,
                        float* __restrict__ ctx, int Lq, int Lk, int u, int causal)
{
    __shared__ float sc[768];
    const int blk = blockIdx.x;
    const int ui  = blk % u;
    const int h   = (blk / u) & (NH - 1);
    const int b   = blk / (u * NH);
    const int lane = threadIdx.x;
    const int row  = Mtop[((size_t)b * NH + h) * u + ui];

    const float* q = Q + ((size_t)b * Lq + row) * DM + h * 16;
    float qv[16];
    #pragma unroll
    for (int d = 0; d < 16; ++d) qv[d] = q[d];

    const float scale = 0.25f; // 1/sqrt(16)
    for (int j = lane; j < Lk; j += 32) {
        const float* kr = Kb + ((size_t)b * Lk + j) * DM + h * 16;
        float dot = 0.f;
        #pragma unroll
        for (int d = 0; d < 16; ++d) dot += qv[d] * kr[d];
        dot *= scale;
        if (causal && j > row) dot = -3.4e38f;
        sc[j] = dot;
    }
    __syncthreads();

    float mx = -3.4e38f;
    for (int j = lane; j < Lk; j += 32) mx = fmaxf(mx, sc[j]);
    for (int o = 16; o > 0; o >>= 1) mx = fmaxf(mx, __shfl_xor(mx, o, 32));

    float ssum = 0.f;
    for (int j = lane; j < Lk; j += 32) {
        float e = __expf(sc[j] - mx);
        sc[j] = e;
        ssum += e;
    }
    for (int o = 16; o > 0; o >>= 1) ssum += __shfl_xor(ssum, o, 32);
    const float inv = 1.f / ssum;
    __syncthreads();

    #pragma unroll
    for (int d = 0; d < 16; ++d) {
        float a = 0.f;
        for (int j = lane; j < Lk; j += 32)
            a += sc[j] * V[((size_t)b * Lk + j) * DM + h * 16 + d];
        for (int o = 16; o > 0; o >>= 1) a += __shfl_xor(a, o, 32);
        if (lane == 0)
            ctx[((size_t)b * Lq + row) * DM + h * 16 + d] = a * inv;
    }
}

// ---------------------------------------------------------------------------
// out = LayerNorm(X (+ Yv)) over D=64; one wave per row, 2 elems per lane.
// ---------------------------------------------------------------------------
__global__ __launch_bounds__(32)
void add_ln_kernel(const float* __restrict__ X, const float* __restrict__ Yv,
                   const float* __restrict__ g, const float* __restrict__ bb,
                   float* __restrict__ out)
{
    const int row = blockIdx.x, lane = threadIdx.x;
    const size_t base = (size_t)row * DM;
    float v0 = X[base + lane], v1 = X[base + lane + 32];
    if (Yv) { v0 += Yv[base + lane]; v1 += Yv[base + lane + 32]; }
    float s = v0 + v1;
    for (int o = 16; o > 0; o >>= 1) s += __shfl_xor(s, o, 32);
    const float mean = s * (1.f / 64.f);
    const float d0 = v0 - mean, d1 = v1 - mean;
    float vs = d0 * d0 + d1 * d1;
    for (int o = 16; o > 0; o >>= 1) vs += __shfl_xor(vs, o, 32);
    const float rstd = rsqrtf(vs * (1.f / 64.f) + 1e-5f);
    out[base + lane]      = d0 * rstd * g[lane] + bb[lane];
    out[base + lane + 32] = d1 * rstd * g[lane + 32] + bb[lane + 32];
}

// Distill helpers ------------------------------------------------------------
__global__ void im2col_kernel(const float* __restrict__ x, float* __restrict__ x2, int L)
{
    int idx = blockIdx.x * blockDim.x + threadIdx.x;
    int total = BATCH * L * 192;
    if (idx >= total) return;
    int col = idx % 192;
    int t   = (idx / 192) % L;
    int b   = idx / (192 * L);
    int i = col / 3, kk = col % 3;
    int ts = t - 1 + kk;
    ts = (ts < 0) ? ts + L : ((ts >= L) ? ts - L : ts);
    x2[idx] = x[((size_t)b * L + ts) * DM + i];
}

__global__ void bn_elu_kernel(float* __restrict__ y, const float* __restrict__ g,
                              const float* __restrict__ b, int total)
{
    int idx = blockIdx.x * blockDim.x + threadIdx.x;
    if (idx >= total) return;
    int c = idx & (DM - 1);
    float v = y[idx] * g[c] + b[c];
    y[idx] = (v > 0.f) ? v : (__expf(v) - 1.f);
}

__global__ void maxpool_kernel(const float* __restrict__ in, float* __restrict__ out, int Lin)
{
    const int Lout = Lin / 2;
    int idx = blockIdx.x * blockDim.x + threadIdx.x;
    if (idx >= BATCH * Lout * DM) return;
    int c = idx & (DM - 1);
    int t = (idx >> 6) % Lout;
    int b = idx / (DM * Lout);
    int w0 = 2 * t - 1;
    float m = -3.4e38f;
    #pragma unroll
    for (int w = 0; w < 3; ++w) {
        int ww = w0 + w;
        if (ww >= 0 && ww < Lin)
            m = fmaxf(m, in[((size_t)b * Lin + ww) * DM + c]);
    }
    out[idx] = m;
}

__global__ void slice_kernel(const float* __restrict__ pr, float* __restrict__ out)
{
    int idx = blockIdx.x * blockDim.x + threadIdx.x;
    if (idx >= BATCH * PRED * CIN) return;
    int c = idx % CIN;
    int t = (idx / CIN) % PRED;
    int b = idx / (CIN * PRED);
    out[idx] = pr[((size_t)b * DEC_LEN + (DEC_LEN - PRED) + t) * CIN + c];
}

// ---------------------------------------------------------------------------
// Host orchestration
// ---------------------------------------------------------------------------
struct AttnWs { float *q, *k, *v, *ctx, *ms, *meanv; int *mtop; };

static void launch_linear(const float* X, const float* W, const float* bias, float* Y,
                          int M, int K, int N, int act, hipStream_t s)
{
    const int mtiles = M / 16;
    int blocks = ceil_div(mtiles, 16);   // ~2 M-tiles per wave, amortize B frags
    if (blocks < 1) blocks = 1;
    if (K == 64 && N <= 16)        gemm_ws_kernel<2, 1><<<blocks, 256, 0, s>>>(X, W, bias, Y, M, N, act);
    else if (K == 64 && N <= 64)   gemm_ws_kernel<2, 4><<<blocks, 256, 0, s>>>(X, W, bias, Y, M, N, act);
    else if (K == 64)              gemm_ws_kernel<2, 8><<<blocks, 256, 0, s>>>(X, W, bias, Y, M, N, act);
    else if (K == 128)             gemm_ws_kernel<4, 4><<<blocks, 256, 0, s>>>(X, W, bias, Y, M, N, act);
    else                           gemm_ws_kernel<6, 4><<<blocks, 256, 0, s>>>(X, W, bias, Y, M, N, act); // K=192
}

static int prob_u(int L)
{
    int u = (int)(3.0 * std::ceil(std::log((double)L)));
    return u < L ? u : L;
}

static void run_attention(const float* qx, int Lq, const float* kvx, int Lk,
                          const float* const* ap /*wq,bq,wk,bk,wv,bv,wo,bo*/,
                          int causal, float* out, const AttnWs& ws, hipStream_t s)
{
    launch_linear(qx,  ap[0], ap[1], ws.q, BATCH * Lq, DM, DM, 0, s);
    launch_linear(kvx, ap[2], ap[3], ws.k, BATCH * Lk, DM, DM, 0, s);
    launch_linear(kvx, ap[4], ap[5], ws.v, BATCH * Lk, DM, DM, 0, s);

    const int Upart = prob_u(Lk);
    const int u     = prob_u(Lq);

    attn_m_kernel<<<ceil_div(BATCH * NH * Lq, 256), 256, 0, s>>>(ws.q, ws.k, ws.ms, Lq, Lk, Upart);
    topk_kernel<<<BATCH * NH, 32, 0, s>>>(ws.ms, ws.mtop, Lq, u);

    if (causal) {
        cumsum_v_kernel<<<ceil_div(BATCH * DM, 256), 256, 0, s>>>(ws.v, ws.ctx, Lk);
    } else {
        mean_v_kernel<<<ceil_div(BATCH * DM, 256), 256, 0, s>>>(ws.v, ws.meanv, Lk);
        fill_mean_kernel<<<ceil_div(BATCH * Lq * DM, 256), 256, 0, s>>>(ws.meanv, ws.ctx, Lq);
    }
    attn_update_kernel<<<BATCH * NH * u, 32, 0, s>>>(ws.q, ws.k, ws.v, ws.mtop, ws.ctx,
                                                     Lq, Lk, u, causal);
    launch_linear(ws.ctx, ap[6], ap[7], out, BATCH * Lq, DM, DM, 0, s);
}

// P: c1_w,c1_b,c2_w,c2_b,n1_g,n1_b,n2_g,n2_b, wq,bq,wk,bk,wv,bv,wo,bo
static void encoder_layer(float* x, int L, const float* const* P,
                          float* ybuf, float* hbuf, const AttnWs& ws, hipStream_t s)
{
    run_attention(x, L, x, L, P + 8, 0, ybuf, ws, s);
    add_ln_kernel<<<BATCH * L, 32, 0, s>>>(x, ybuf, P[4], P[5], x);
    launch_linear(x, P[0], P[1], hbuf, BATCH * L, DM, DFF, 1, s);   // GELU
    launch_linear(hbuf, P[2], P[3], ybuf, BATCH * L, DFF, DM, 0, s);
    add_ln_kernel<<<BATCH * L, 32, 0, s>>>(x, ybuf, P[6], P[7], x);
}

extern "C" void kernel_launch(void* const* d_in, const int* in_sizes, int n_in,
                              void* d_out, int out_size, void* d_ws, size_t ws_size,
                              hipStream_t stream)
{
    (void)in_sizes; (void)n_in; (void)out_size; (void)ws_size;
    const float* const* in = (const float* const*)d_in;

    // --- input pointer map (setup_inputs insertion order, params recursed) ---
    const float* x_enc      = in[0];
    const float* x_mark_enc = in[1];
    const float* x_dec      = in[2];
    const float* x_mark_dec = in[3];
    const float* enc_tokw = in[4], *enc_timw = in[5];
    const float* dec_tokw = in[6], *dec_timw = in[7];
    const float* encL0[16], *encL1[16];
    for (int i = 0; i < 16; ++i) { encL0[i] = in[8 + i]; encL1[i] = in[24 + i]; }
    const float* dist_w = in[40], *dist_b = in[41], *bn_g = in[42], *bn_b = in[43];
    const float* enc_ng = in[44], *enc_nb = in[45];
    const float* dc1w = in[46], *dc1b = in[47], *dc2w = in[48], *dc2b = in[49];
    const float* dn1g = in[50], *dn1b = in[51], *dn2g = in[52], *dn2b = in[53];
    const float* dn3g = in[54], *dn3b = in[55];
    const float* selfA[8], *crossA[8];
    for (int i = 0; i < 8; ++i) { selfA[i] = in[56 + i]; crossA[i] = in[64 + i]; }
    const float* dec_ng = in[72], *dec_nb = in[73];
    const float* proj_w = in[74], *proj_b = in[75];

    // --- workspace bump allocator ---
    char* wsp = (char*)d_ws;
    size_t off = 0;
    auto alloc = [&](size_t nbytes) -> void* {
        void* p = wsp + off;
        off += (nbytes + 255) & ~(size_t)255;
        return p;
    };
    const size_t TOK = (size_t)BATCH * SEQ_LEN;            // max tokens
    float* xbuf   = (float*)alloc(TOK * DM  * 4);
    float* ybuf   = (float*)alloc(TOK * DM  * 4);
    float* hbuf   = (float*)alloc(TOK * 192 * 4);          // FFN hidden / im2col
    float* encout = (float*)alloc((size_t)BATCH * ENC_L2 * DM * 4);
    float* projb  = (float*)alloc((size_t)BATCH * DEC_LEN * CIN * 4);
    AttnWs ws;
    ws.q     = (float*)alloc(TOK * DM * 4);
    ws.k     = (float*)alloc(TOK * DM * 4);
    ws.v     = (float*)alloc(TOK * DM * 4);
    ws.ctx   = (float*)alloc(TOK * DM * 4);
    ws.ms    = (float*)alloc((size_t)BATCH * NH * SEQ_LEN * 4);
    ws.meanv = (float*)alloc((size_t)BATCH * DM * 4);
    ws.mtop  = (int*)  alloc((size_t)BATCH * NH * 32 * 4);

    // ================= Encoder =================
    embed_kernel<<<ceil_div(BATCH * SEQ_LEN * DM, 256), 256, 0, stream>>>(
        x_enc, x_mark_enc, enc_tokw, enc_timw, xbuf, SEQ_LEN, CIN);

    encoder_layer(xbuf, SEQ_LEN, encL0, ybuf, hbuf, ws, stream);

    // distill: circular conv (im2col + WMMA GEMM, K=192) -> BN scale -> ELU -> maxpool/2
    im2col_kernel<<<ceil_div(BATCH * SEQ_LEN * 192, 256), 256, 0, stream>>>(xbuf, hbuf, SEQ_LEN);
    launch_linear(hbuf, dist_w, dist_b, ybuf, BATCH * SEQ_LEN, 192, DM, 0, stream);
    bn_elu_kernel<<<ceil_div(BATCH * SEQ_LEN * DM, 256), 256, 0, stream>>>(
        ybuf, bn_g, bn_b, BATCH * SEQ_LEN * DM);
    maxpool_kernel<<<ceil_div(BATCH * ENC_L2 * DM, 256), 256, 0, stream>>>(ybuf, xbuf, SEQ_LEN);

    encoder_layer(xbuf, ENC_L2, encL1, ybuf, hbuf, ws, stream);
    add_ln_kernel<<<BATCH * ENC_L2, 32, 0, stream>>>(xbuf, nullptr, enc_ng, enc_nb, encout);

    // ================= Decoder =================
    embed_kernel<<<ceil_div(BATCH * DEC_LEN * DM, 256), 256, 0, stream>>>(
        x_dec, x_mark_dec, dec_tokw, dec_timw, xbuf, DEC_LEN, CIN);

    // masked ProbSparse self-attention + LN
    run_attention(xbuf, DEC_LEN, xbuf, DEC_LEN, selfA, 1, ybuf, ws, stream);
    add_ln_kernel<<<BATCH * DEC_LEN, 32, 0, stream>>>(xbuf, ybuf, dn1g, dn1b, xbuf);

    // cross-attention vs encoder output + LN
    run_attention(xbuf, DEC_LEN, encout, ENC_L2, crossA, 0, ybuf, ws, stream);
    add_ln_kernel<<<BATCH * DEC_LEN, 32, 0, stream>>>(xbuf, ybuf, dn2g, dn2b, xbuf);

    // FFN + LN(n3) + final decoder LN
    launch_linear(xbuf, dc1w, dc1b, hbuf, BATCH * DEC_LEN, DM, DFF, 1, stream);
    launch_linear(hbuf, dc2w, dc2b, ybuf, BATCH * DEC_LEN, DFF, DM, 0, stream);
    add_ln_kernel<<<BATCH * DEC_LEN, 32, 0, stream>>>(xbuf, ybuf, dn3g, dn3b, xbuf);
    add_ln_kernel<<<BATCH * DEC_LEN, 32, 0, stream>>>(xbuf, nullptr, dec_ng, dec_nb, xbuf);

    // projection (N=7 via WMMA tile with column guard) + tail slice
    launch_linear(xbuf, proj_w, proj_b, projb, BATCH * DEC_LEN, DM, CIN, 0, stream);
    slice_kernel<<<ceil_div(BATCH * PRED * CIN, 256), 256, 0, stream>>>(projb, (float*)d_out);
}
